// SOM_59244778881720
// MI455X (gfx1250) — matrix operand order repaired and checked by reading, not compiled, not run on hardware
//
#include <hip/hip_runtime.h>
#include <hip/hip_bf16.h>
#include <stdint.h>

// SOM BMU search: dists = ||x[:,None]-w+eps|| -> per-row min + argmin + mean-min loss.
// dist^2(m,n) = x2[m] + w2[n] - 2*xw[m,n] + 2*eps*(sx[m]-sw[n]) + D*eps^2
// xw GEMM via bf16x3 split-precision V_WMMA_F32_16X16X32_BF16.
// w pre-split/transposed ONCE to bf16 hi/lo planes [K][D]; GEMM stages them with
// GLOBAL_LOAD_ASYNC_TO_LDS_B128 double-buffering (ASYNCcnt pipelined with WMMA).

#define B_ROWS 8192
#define DIM    1024
#define KCOLS  4096
#define EPSV   1e-6f

#define BM 64                 // rows per block
#define BN 64                 // cols per n-tile iteration
#define KC 64                 // d-chunk of w staged per pipeline stage
#define NCHUNK (DIM / KC)     // 16
#define A_STRIDE (DIM + 8)    // bf16 elems; row stride 2064 B (16B aligned, padded)
#define B_STRIDE (KC + 8)     // bf16 elems; row stride 144 B (16B aligned, padded)

typedef __attribute__((ext_vector_type(16))) __bf16 v16bf;
typedef __attribute__((ext_vector_type(8)))  __bf16 v8bf;
typedef __attribute__((ext_vector_type(4)))  __bf16 v4bf;
typedef __attribute__((ext_vector_type(2)))  __bf16 v2bf;
typedef __attribute__((ext_vector_type(8)))  float  v8f;
typedef __attribute__((ext_vector_type(4)))  int    v4i;

#if defined(__HIP_DEVICE_COMPILE__) && \
    __has_builtin(__builtin_amdgcn_global_load_async_to_lds_b128) && \
    __has_builtin(__builtin_amdgcn_s_wait_asynccnt)
#define USE_ASYNC_LDS 1
#define ASYNC_WAIT(n) __builtin_amdgcn_s_wait_asynccnt(n)
#else
#define USE_ASYNC_LDS 0
#define ASYNC_WAIT(n)
#endif

// A fragment (16x32 bf16): per lane K offsets {0..7, 16..23} (lanes>=16 shifted +8)
__device__ __forceinline__ v16bf ld_a_frag(const __bf16* p) {
  v8bf a = *(const v8bf*)p;          // K +0..7
  v8bf b = *(const v8bf*)(p + 16);   // K +16..23
  v16bf r;
#pragma unroll
  for (int i = 0; i < 8; ++i) { r[i] = a[i]; r[i + 8] = b[i]; }
  return r;
}

// B fragment (32x16 bf16): per lane K offsets {0..7, 16..23} (lanes>=16 shifted +8)
__device__ __forceinline__ v16bf ld_b_frag(const __bf16* p) {
  v8bf a = *(const v8bf*)p;
  v8bf b = *(const v8bf*)(p + 16);
  v16bf r;
#pragma unroll
  for (int i = 0; i < 8; ++i) { r[i] = a[i]; r[i + 8] = b[i]; }
  return r;
}

// ---------------- per-row stats: x2, sx ----------------
__global__ void __launch_bounds__(256)
som_row_stats(const float* __restrict__ x, float* __restrict__ rs) {
  const int lane = threadIdx.x & 31;
  const int wid  = threadIdx.x >> 5;
  const int row  = blockIdx.x * 8 + wid;
  const float* xr = x + (size_t)row * DIM;
  float s2 = 0.f, s1 = 0.f;
  for (int i = lane; i < DIM; i += 32) { float v = xr[i]; s2 += v * v; s1 += v; }
#pragma unroll
  for (int off = 16; off > 0; off >>= 1) {
    s2 += __shfl_down(s2, off);
    s1 += __shfl_down(s1, off);
  }
  if (lane == 0) { rs[row * 2 + 0] = s2; rs[row * 2 + 1] = s1; }
}

// ---------------- per-col stats: w2, sw ----------------
__global__ void __launch_bounds__(256)
som_col_stats(const float* __restrict__ w, float* __restrict__ cs) {
  const int k = blockIdx.x * 256 + threadIdx.x;
  float s2 = 0.f, s1 = 0.f;
  for (int d = 0; d < DIM; ++d) {          // coalesced across threads
    float v = w[(size_t)d * KCOLS + k];
    s2 += v * v; s1 += v;
  }
  cs[k * 2 + 0] = s2;
  cs[k * 2 + 1] = s1;
}

// ---------------- one-time w split+transpose: w[d][n] fp32 -> Whi_t/Wlo_t[n][d] bf16 ----------------
__global__ void __launch_bounds__(256)
som_w_split(const float* __restrict__ w, __bf16* __restrict__ whit, __bf16* __restrict__ wlot) {
  __shared__ __bf16 Thi[64][64 + 2];
  __shared__ __bf16 Tlo[64][64 + 2];
  const int dt = blockIdx.x >> 6;          // DIM/64 = 16 d-tiles
  const int nt = blockIdx.x & 63;          // KCOLS/64 = 64 n-tiles
  const int d0 = dt * 64, n0 = nt * 64;
  const int tid = threadIdx.x;
#pragma unroll
  for (int i = 0; i < 16; ++i) {           // 64x64 tile, coalesced in n
    int e  = tid + i * 256;
    int d  = e >> 6;
    int nn = e & 63;
    float v = w[(size_t)(d0 + d) * KCOLS + n0 + nn];
    __bf16 h = (__bf16)v;
    Thi[nn][d] = h;
    Tlo[nn][d] = (__bf16)(v - (float)h);
  }
  __syncthreads();
#pragma unroll
  for (int i = 0; i < 8; ++i) {            // write transposed, coalesced in d, 4B stores
    int e  = tid + i * 256;
    int nn = e >> 5;
    int d2 = e & 31;
    size_t off = (size_t)(n0 + nn) * DIM + d0 + d2 * 2;
    v2bf hv = {Thi[nn][d2 * 2], Thi[nn][d2 * 2 + 1]};
    v2bf lv = {Tlo[nn][d2 * 2], Tlo[nn][d2 * 2 + 1]};
    *(v2bf*)(whit + off) = hv;
    *(v2bf*)(wlot + off) = lv;
  }
}

// Stage one w chunk (hi+lo planes, [BN][KC] bf16) into an LDS double buffer.
// 2 planes * 64 rows * 8 sixteen-byte chunks = 1024 issues -> 4 per thread.
__device__ __forceinline__ void stage_w_chunk(const __bf16* __restrict__ whit,
                                              const __bf16* __restrict__ wlot,
                                              __bf16* __restrict__ Bb,   // [2][BN][B_STRIDE]
                                              int n0, int dc, int tid) {
#pragma unroll
  for (int i = 0; i < 4; ++i) {
    int cidx  = tid + i * 256;
    int plane = cidx >> 9;
    int rowc  = cidx & 511;
    int nn    = rowc >> 3;
    int c     = rowc & 7;
    const __bf16* g = (plane ? wlot : whit) + (size_t)(n0 + nn) * DIM + dc + c * 8;
    __bf16* l = Bb + ((size_t)plane * BN + nn) * B_STRIDE + c * 8;
#if USE_ASYNC_LDS
    __builtin_amdgcn_global_load_async_to_lds_b128(
        (v4i __attribute__((address_space(1))) *)(void*)g,
        (v4i __attribute__((address_space(3))) *)(void*)l, 0, 0);
#else
    *(v8bf*)l = *(const v8bf*)g;
#endif
  }
}

// ---------------- fused GEMM + distance + min/argmin ----------------
__global__ void __launch_bounds__(256)
som_gemm_fused(const float* __restrict__ x,
               const __bf16* __restrict__ whit, const __bf16* __restrict__ wlot,
               const float* __restrict__ rs, const float* __restrict__ cs,
               float* __restrict__ out_idx, float* __restrict__ row_min_d) {
  extern __shared__ char smem[];
  __bf16* Ahi = (__bf16*)smem;                          // [BM][A_STRIDE]
  __bf16* Alo = Ahi + (size_t)BM * A_STRIDE;            // [BM][A_STRIDE]
  __bf16* Bb0 = Alo + (size_t)BM * A_STRIDE;            // [2][BN][B_STRIDE] buffer 0
  __bf16* Bb1 = Bb0 + (size_t)2 * BN * B_STRIDE;        // [2][BN][B_STRIDE] buffer 1
  unsigned long long* rowmin = (unsigned long long*)(Bb1 + (size_t)2 * BN * B_STRIDE);
  float* x2l = (float*)(rowmin + BM);
  float* sxl = x2l + BM;

  const int tid  = threadIdx.x;
  const int lane = tid & 31;
  const int wid  = tid >> 5;       // 8 waves
  const int wm   = wid & 3;        // 4 row tiles of 16
  const int wn2  = wid >> 2;       // 2 col half-stripes of 32
  const int row0 = blockIdx.x * BM;
  const bool hiHalf = (lane & 16) != 0;

  // Stage the 64x1024 x stripe pre-split into bf16 hi/lo planes (264 KB LDS), once.
#pragma unroll
  for (int i = 0; i < (BM * DIM / 4) / 256; ++i) {
    int idx4 = tid + i * 256;
    int r  = idx4 >> 8;
    int c4 = idx4 & 255;
    float4 v = *(const float4*)(x + (size_t)(row0 + r) * DIM + c4 * 4);
    __bf16 h0 = (__bf16)v.x, h1 = (__bf16)v.y, h2 = (__bf16)v.z, h3 = (__bf16)v.w;
    v4bf hi = {h0, h1, h2, h3};
    v4bf lo = {(__bf16)(v.x - (float)h0), (__bf16)(v.y - (float)h1),
               (__bf16)(v.z - (float)h2), (__bf16)(v.w - (float)h3)};
    *(v4bf*)(Ahi + (size_t)r * A_STRIDE + c4 * 4) = hi;
    *(v4bf*)(Alo + (size_t)r * A_STRIDE + c4 * 4) = lo;
  }
  if (tid < BM) {
    rowmin[tid] = ~0ull;
    x2l[tid] = rs[(row0 + tid) * 2 + 0];
    sxl[tid] = rs[(row0 + tid) * 2 + 1];
  }
  __syncthreads();

  const int mloc = wm * 16 + (lane & 15);
  const int nl0  = wn2 * 32 + (lane & 15);
  const int nl1  = nl0 + 16;

  for (int n0 = 0; n0 < KCOLS; n0 += BN) {
    v8f acc0 = {}; v8f acc1 = {};

    // Pipeline prologue: chunks 0 and 1 in flight.
    stage_w_chunk(whit, wlot, Bb0, n0, 0, tid);
    stage_w_chunk(whit, wlot, Bb1, n0, KC, tid);

    for (int c = 0; c < NCHUNK; ++c) {
      if (c + 1 < NCHUNK) { ASYNC_WAIT(4); }   // chunk c arrived; c+1 stays in flight
      else               { ASYNC_WAIT(0); }
      __syncthreads();                          // all waves' chunk-c data visible

      const __bf16* Bhi = (c & 1) ? Bb1 : Bb0;
      const __bf16* Blo = Bhi + (size_t)BN * B_STRIDE;
      const int dc = c * KC;

#pragma unroll
      for (int ks = 0; ks < KC / 32; ++ks) {
        const int kA = dc + ks * 32 + (hiHalf ? 8 : 0);
        v16bf ahi = ld_a_frag(Ahi + (size_t)mloc * A_STRIDE + kA);
        v16bf alo = ld_a_frag(Alo + (size_t)mloc * A_STRIDE + kA);

        const int kB = ks * 32 + (hiHalf ? 8 : 0);
        v16bf bhi0 = ld_b_frag(Bhi + nl0 * B_STRIDE + kB);
        v16bf blo0 = ld_b_frag(Blo + nl0 * B_STRIDE + kB);
        v16bf bhi1 = ld_b_frag(Bhi + nl1 * B_STRIDE + kB);
        v16bf blo1 = ld_b_frag(Blo + nl1 * B_STRIDE + kB);

        // bf16x3: hi*hi + hi*lo + lo*hi into fp32 accumulators
        acc0 = __builtin_amdgcn_wmma_f32_16x16x32_bf16(false, ahi, false, bhi0, (short)0, acc0, false, false);
        acc1 = __builtin_amdgcn_wmma_f32_16x16x32_bf16(false, ahi, false, bhi1, (short)0, acc1, false, false);
        acc0 = __builtin_amdgcn_wmma_f32_16x16x32_bf16(false, ahi, false, blo0, (short)0, acc0, false, false);
        acc1 = __builtin_amdgcn_wmma_f32_16x16x32_bf16(false, ahi, false, blo1, (short)0, acc1, false, false);
        acc0 = __builtin_amdgcn_wmma_f32_16x16x32_bf16(false, alo, false, bhi0, (short)0, acc0, false, false);
        acc1 = __builtin_amdgcn_wmma_f32_16x16x32_bf16(false, alo, false, bhi1, (short)0, acc1, false, false);
      }

      __syncthreads();                          // everyone done reading buf[c&1]
      if (c + 2 < NCHUNK) {                     // refill the freed buffer
        stage_w_chunk(whit, wlot, (c & 1) ? Bb1 : Bb0, n0, (c + 2) * KC, tid);
      }
    }

    // Epilogue: dist + packed (dist_bits<<32 | col) ds_min_u64 per row.
    // Non-negative float bits are monotonic as uint; ties pick smaller col
    // (matches jnp.argmin first-occurrence). Min is order-independent.
    {
      const int ng0 = n0 + nl0;
      const int ng1 = n0 + nl1;
      const float w2c0 = cs[ng0 * 2 + 0], swc0 = cs[ng0 * 2 + 1];
      const float w2c1 = cs[ng1 * 2 + 0], swc1 = cs[ng1 * 2 + 1];
      const int mbase = wm * 16 + (hiHalf ? 8 : 0);
#pragma unroll
      for (int j = 0; j < 8; ++j) {
        const int ml = mbase + j;
        const float base = x2l[ml] + 2.0f * EPSV * sxl[ml] + (float)DIM * (EPSV * EPSV);
        float sq0 = base + w2c0 - 2.0f * acc0[j] - 2.0f * EPSV * swc0;
        float d0  = __builtin_sqrtf(fmaxf(sq0, 0.0f));
        atomicMin(&rowmin[ml],
                  ((unsigned long long)__float_as_uint(d0) << 32) | (unsigned)ng0);
        float sq1 = base + w2c1 - 2.0f * acc1[j] - 2.0f * EPSV * swc1;
        float d1  = __builtin_sqrtf(fmaxf(sq1, 0.0f));
        atomicMin(&rowmin[ml],
                  ((unsigned long long)__float_as_uint(d1) << 32) | (unsigned)ng1);
      }
    }
  }

  __syncthreads();
  if (tid < BM) {
    unsigned long long v = rowmin[tid];
    out_idx[row0 + tid]   = (float)(unsigned)(v & 0xffffffffu);
    row_min_d[row0 + tid] = __uint_as_float((unsigned)(v >> 32));
  }
}

// ---------------- deterministic fixed-order loss reduction ----------------
__global__ void __launch_bounds__(256)
som_loss(const float* __restrict__ row_min_d, float* __restrict__ out0) {
  __shared__ float sm[256];
  float s = 0.f;
  for (int i = threadIdx.x; i < B_ROWS; i += 256) s += row_min_d[i];
  sm[threadIdx.x] = s;
  __syncthreads();
  for (int off = 128; off > 0; off >>= 1) {
    if (threadIdx.x < off) sm[threadIdx.x] += sm[threadIdx.x + off];
    __syncthreads();
  }
  if (threadIdx.x == 0) out0[0] = sm[0] / (float)B_ROWS;
}

extern "C" void kernel_launch(void* const* d_in, const int* in_sizes, int n_in,
                              void* d_out, int out_size, void* d_ws, size_t ws_size,
                              hipStream_t stream) {
  (void)in_sizes; (void)n_in; (void)out_size; (void)ws_size;
  const float* x = (const float*)d_in[0];       // [8192,1024]
  const float* w = (const float*)d_in[1];       // [1024,4096]
  float* out = (float*)d_out;                   // [0]=loss, [1..8192]=bmu index

  float*  rs       = (float*)d_ws;              // [8192][2] x2,sx
  float*  cs       = rs + 2 * B_ROWS;           // [4096][2] w2,sw
  float*  row_mind = cs + 2 * KCOLS;            // [8192] min dist per row
  __bf16* whit     = (__bf16*)(row_mind + B_ROWS);        // [4096][1024] bf16 hi
  __bf16* wlot     = whit + (size_t)KCOLS * DIM;          // [4096][1024] bf16 lo

  const size_t lds_bytes = (size_t)2 * BM * A_STRIDE * 2      // x hi/lo bf16 planes
                         + (size_t)2 * 2 * BN * B_STRIDE * 2  // w hi/lo double buffers
                         + (size_t)BM * 8                     // rowmin u64
                         + (size_t)2 * BM * 4;                // x2,sx
  static int lds_set = 0;  // idempotent attribute set (same value every call)
  if (!lds_set) {
    (void)hipFuncSetAttribute((const void*)som_gemm_fused,
                              hipFuncAttributeMaxDynamicSharedMemorySize,
                              (int)lds_bytes);
    lds_set = 1;
  }

  som_row_stats<<<B_ROWS / 8, 256, 0, stream>>>(x, rs);
  som_col_stats<<<KCOLS / 256, 256, 0, stream>>>(w, cs);
  som_w_split<<<(DIM / 64) * (KCOLS / 64), 256, 0, stream>>>(w, whit, wlot);
  som_gemm_fused<<<B_ROWS / BM, 256, lds_bytes, stream>>>(x, whit, wlot, rs, cs, out + 1, row_mind);
  som_loss<<<1, 256, 0, stream>>>(row_mind, out);
}
